// Transducer_17763984736854
// MI455X (gfx1250) — matrix operand order repaired and checked by reading, not compiled, not run on hardware
//
#include <hip/hip_runtime.h>
#include <hip/hip_bf16.h>
#include <math.h>

// ---------------- problem constants (from setup_inputs) ----------------
#define Bn   8
#define Tn   512
#define T1n  513
#define Sn   128
#define Sp1  129
#define Cn   512
#define Rn   19          // prune range
#define NEGF (-1e20f)
#define LMW  0.25f       // lm_only_scale
#define CMW  0.75f       // 1 - lm_only_scale

typedef __attribute__((ext_vector_type(8)))  float  v8f;
typedef __attribute__((ext_vector_type(16))) __bf16 bf16x16;
typedef __attribute__((ext_vector_type(8)))  __bf16 bf16x8;

union ABu { bf16x16 v; bf16x8 h[2]; };

__device__ __forceinline__ v8f wmma_bf16(bf16x16 a, bf16x16 b, v8f c) {
  // D = A(16x32 bf16) x B(32x16 bf16) + C(16x16 f32)
  return __builtin_amdgcn_wmma_f32_16x16x32_bf16(false, a, false, b,
                                                 (short)0, c, false, false);
}

__device__ __forceinline__ float lae(float a, float b) {
  float m = fmaxf(a, b), n = fminf(a, b);
  return m + log1pf(__expf(n - m));
}

// ---------------- K1: per-row max / exp / (optional) logsumexp ----------------
__global__ void rowstats_kernel(const float* __restrict__ x, __bf16* __restrict__ xe,
                                float* __restrict__ mx_out, float* __restrict__ lse_out) {
  __shared__ float red[256];
  int row = blockIdx.x, tid = threadIdx.x;
  const float* xr = x + (size_t)row * Cn;
  __bf16* er = xe + (size_t)row * Cn;
  float m = -3.0e38f;
  for (int k = tid; k < Cn; k += 256) m = fmaxf(m, xr[k]);
  red[tid] = m; __syncthreads();
  for (int off = 128; off > 0; off >>= 1) {
    if (tid < off) red[tid] = fmaxf(red[tid], red[tid + off]);
    __syncthreads();
  }
  float mx = red[0]; __syncthreads();
  float s = 0.f;
  for (int k = tid; k < Cn; k += 256) {
    float e = __expf(xr[k] - mx);
    er[k] = (__bf16)e;
    s += e;
  }
  red[tid] = s; __syncthreads();
  for (int off = 128; off > 0; off >>= 1) {
    if (tid < off) red[tid] += red[tid + off];
    __syncthreads();
  }
  if (tid == 0) {
    mx_out[row] = mx;
    if (lse_out) lse_out[row] = mx + logf(red[0]);
  }
}

// ---------------- K2: transpose joiner_w (k,d) -> Wt (d,k) bf16 ----------------
__global__ void wtprep_kernel(const float* __restrict__ W, __bf16* __restrict__ Wt) {
  int d = blockIdx.x, tid = threadIdx.x;
  for (int k = tid; k < Cn; k += 256)
    Wt[(size_t)d * Cn + k] = (__bf16)W[(size_t)k * Cn + d];
}

// ---------------- K3: normalizers[b,s,t] = log(lm_e . am_e) + maxes (WMMA) ----------------
// 9 s-tiles (129 rows padded to 144), 32 t-tiles, per batch. 8 waves/block, 1 tile/wave.
__global__ void __launch_bounds__(256)
normalizer_kernel(const __bf16* __restrict__ lm_e, const __bf16* __restrict__ am_e,
                  const float* __restrict__ lm_max, const float* __restrict__ am_max,
                  float* __restrict__ nrm) {
  int wv = threadIdx.x >> 5, lane = threadIdx.x & 31;
  int half = lane >> 4, l16 = lane & 15;
  int tile = blockIdx.x * 8 + wv;                 // 0 .. 2303
  int b = tile / (9 * 32); int rm = tile % (9 * 32);
  int st = rm / 32, tt = rm % 32;
  int s0 = st * 16, t0 = tt * 16;
  int sA = s0 + l16; if (sA > Sn) sA = Sn;        // clamp pad rows
  const __bf16* Ab = lm_e + (size_t)(b * Sp1 + sA) * Cn;
  const __bf16* Bb = am_e + (size_t)(b * Tn + t0 + l16) * Cn;
  v8f acc = {0.f, 0.f, 0.f, 0.f, 0.f, 0.f, 0.f, 0.f};
  for (int kt = 0; kt < 16; ++kt) {
    int kb = kt * 32;
    ABu a;
    a.h[0] = *(const bf16x8*)&Ab[kb + half * 16];
    a.h[1] = *(const bf16x8*)&Ab[kb + half * 16 + 8];
    bf16x16 bv = *(const bf16x16*)&Bb[kb + half * 16];
    acc = wmma_bf16(a.v, bv, acc);
  }
  int tcol = t0 + l16;
  float amx = am_max[b * Tn + tcol];
#pragma unroll
  for (int i = 0; i < 8; ++i) {
    int s = s0 + half * 8 + i;
    if (s <= Sn)
      nrm[(size_t)(b * Sp1 + s) * Tn + tcol] =
          logf(acc[i]) + lm_max[b * Sp1 + s] + amx;
  }
}

// ---------------- K4: smoothed px/py ----------------
__global__ void smoothed_kernel(const float* __restrict__ am, const float* __restrict__ lm,
                                const int* __restrict__ targets,
                                const float* __restrict__ nrm, const float* __restrict__ lm_lse,
                                float* __restrict__ px1, float* __restrict__ py1) {
  int blk = blockIdx.x, tid = threadIdx.x;
  int b = blk / Sp1, s = blk % Sp1;
  float lm0 = lm[(size_t)(b * Sp1 + s) * Cn + 0];
  float lse = lm_lse[b * Sp1 + s];
  int sym = (s < Sn) ? targets[b * Sn + s] : 0;
  float lms = lm[(size_t)(b * Sp1 + s) * Cn + sym];
  for (int t = tid; t < Tn; t += 256) {
    float nv = nrm[(size_t)(b * Sp1 + s) * Tn + t];
    float py = am[(size_t)(b * Tn + t) * Cn + 0] + lm0 - nv;
    py1[(size_t)(b * Sp1 + s) * Tn + t] = CMW * py + LMW * (lm0 - lse);
    if (s < Sn) {
      float px = am[(size_t)(b * Tn + t) * Cn + sym] + lms - nv;
      px1[(size_t)(b * Sn + s) * T1n + t] = CMW * px + LMW * (lms - lse);
    }
  }
  if (tid == 0 && s < Sn) px1[(size_t)(b * Sn + s) * T1n + Tn] = NEGF;
}

// ---------------- K5: mutual-information forward DP (anti-diagonal wavefront) ----------------
__global__ void mi_forward_kernel(const float* __restrict__ px, const float* __restrict__ py,
                                  float* __restrict__ p, float* __restrict__ plast) {
  int b = blockIdx.x, tid = threadIdx.x;
  const float* pxb = px + (size_t)b * Sn * T1n;
  const float* pyb = py + (size_t)b * Sp1 * Tn;
  float* pb = p + (size_t)b * Sp1 * T1n;
  if (tid == 0) {                          // row s = 0: cumsum of py row 0
    float c = 0.f; pb[0] = 0.f;
    for (int t = 1; t <= Tn; ++t) { c += pyb[t - 1]; pb[t] = c; }
  }
  __threadfence_block(); __syncthreads();
  int s = tid;
  for (int d = 1; d <= Sn + Tn; ++d) {
    if (s >= 1 && s <= Sn) {
      int t = d - s;
      if (t >= 0 && t <= Tn) {
        float a = pb[(s - 1) * T1n + t] + pxb[(s - 1) * T1n + t];
        float c = (t > 0) ? pb[s * T1n + (t - 1)] + pyb[s * Tn + (t - 1)] : NEGF;
        pb[s * T1n + t] = lae(a, c);
      }
    }
    __threadfence_block(); __syncthreads();
  }
  if (tid == 0) plast[b] = pb[Sn * T1n + Tn];
}

// ---------------- K6: backward DP + occupancy gradients ----------------
__global__ void mi_backward_kernel(const float* __restrict__ px, const float* __restrict__ py,
                                   const float* __restrict__ p, const float* __restrict__ plast,
                                   float* __restrict__ q, float* __restrict__ gpx,
                                   float* __restrict__ gpy) {
  int b = blockIdx.x, s = threadIdx.x;
  const float* pxb = px + (size_t)b * Sn * T1n;
  const float* pyb = py + (size_t)b * Sp1 * Tn;
  const float* pb  = p  + (size_t)b * Sp1 * T1n;
  float* qb  = q   + (size_t)b * Sp1 * T1n;
  float* gxb = gpx + (size_t)b * Sn * T1n;
  float* gyb = gpy + (size_t)b * Sp1 * Tn;
  for (int d = Sn + Tn; d >= 0; --d) {
    if (s <= Sn) {
      int t = d - s;
      if (t >= 0 && t <= Tn) {
        if (s == Sn && t == Tn) qb[s * T1n + t] = 0.f;
        else {
          float a = (s < Sn) ? pxb[s * T1n + t] + qb[(s + 1) * T1n + t] : NEGF;
          float c = (t < Tn) ? pyb[s * Tn + t] + qb[s * T1n + (t + 1)] : NEGF;
          qb[s * T1n + t] = lae(a, c);
        }
      }
    }
    __threadfence_block(); __syncthreads();
  }
  float tot = plast[b];
  if (s <= Sn) {
    if (s < Sn)
      for (int t = 0; t <= Tn; ++t)
        gxb[s * T1n + t] = __expf(pb[s * T1n + t] + pxb[s * T1n + t] +
                                  qb[(s + 1) * T1n + t] - tot);
    for (int t = 0; t < Tn; ++t)
      gyb[s * Tn + t] = __expf(pb[s * T1n + t] + pyb[s * Tn + t] +
                               qb[s * T1n + (t + 1)] - tot);
  }
}

// ---------------- K7: windowed-occupancy argmax -> raw s_begin ----------------
__global__ void sbegin_kernel(const float* __restrict__ gpx, const float* __restrict__ gpy,
                              int* __restrict__ sbr) {
  int idx = blockIdx.x * 256 + threadIdx.x;   // 0..4095 = (b,t)
  if (idx >= Bn * Tn) return;
  int b = idx >> 9, t = idx & 511;
  const float* gx = gpx + (size_t)b * Sn * T1n;
  const float* gy = gpy + (size_t)b * Sp1 * Tn;
  float ring[20];
  float win = 0.f;
  for (int j = 1; j <= Rn; ++j) {             // tot rows 1..19 (row j -> s=j-1)
    int s = j - 1;
    float v = ((s < Sn) ? gx[s * T1n + t] : 0.f) + gy[s * Tn + t];
    ring[j % 20] = v; win += v;
  }
  int bw = 0; float best = win;
  for (int w2 = 1; w2 <= (Sp1 + 1 - Rn) - 1; ++w2) {   // windows 1..110
    int s = w2 + Rn - 1;                               // new row s = w2+18 <= 128
    float v = ((s < Sn) ? gx[s * T1n + t] : 0.f) + gy[s * Tn + t];
    win += v - ring[w2 % 20];
    ring[(w2 + Rn) % 20] = v;
    if (win > best) { best = win; bw = w2; }
  }
  sbr[idx] = bw;
}

// ---------------- K8: monotonic lower-bound adjust (sequential per batch) ----------------
__global__ void adjust_kernel(int* __restrict__ sb) {
  int b = threadIdx.x;
  if (b >= Bn) return;
  int* r = sb + b * Tn;
  int mn = 1 << 30;
  for (int t = Tn - 1; t >= 0; --t) { mn = min(mn, r[t]); r[t] = mn; }
  int ym = 1 << 30;
  for (int t = Tn - 1; t >= 0; --t) {
    int y = (Rn - 1) * t - r[t];
    ym = min(ym, y);
    int yc = max(ym, 0);
    r[t] = (Rn - 1) * t - yc;
  }
}

// ---------------- K9: fused joiner GEMM (bf16 WMMA) + bias + logsumexp + gathers ----------------
// One workgroup per (b,t): 32x512 logits (rows 0..18 valid) never leave registers/LDS.
__global__ void __launch_bounds__(256)
joiner_kernel(const float* __restrict__ am, const float* __restrict__ lm,
              const int* __restrict__ targets, const __bf16* __restrict__ Wt,
              const float* __restrict__ jbias, const int* __restrict__ sbeg,
              float* __restrict__ pxband, float* __restrict__ pyband) {
  __shared__ __align__(16) __bf16 Xs[32][Cn];   // tanh(am + lm_pruned), bf16
  __shared__ float amr[Cn];
  __shared__ int   symc[32];
  __shared__ float pxl[32], pyl[32];
  __shared__ float wmx[8][32], wsm[8][32];

  int tid = threadIdx.x;
  int bt = blockIdx.x, b = bt >> 9, t = bt & 511;
  int sb = sbeg[bt];

  for (int k = tid; k < Cn; k += 256) amr[k] = am[(size_t)(b * Tn + t) * Cn + k];
  if (tid < 32) {
    int s = sb + tid;
    symc[tid] = (tid < Rn && s < Sn) ? targets[b * Sn + s] : 0;  // sym_t (blank at s==S)
  }
  __syncthreads();

  for (int it = tid; it < 32 * Cn; it += 256) {
    int m = it >> 9, k = it & 511;
    float v = 0.f;
    if (m < Rn) {
      int s = sb + m;
      v = tanhf(amr[k] + lm[(size_t)(b * Sp1 + s) * Cn + k]);
    }
    Xs[m][k] = (__bf16)v;
  }
  __syncthreads();

  int w = tid >> 5, lane = tid & 31, half = lane >> 4, l16 = lane & 15;
  v8f z = {0.f, 0.f, 0.f, 0.f, 0.f, 0.f, 0.f, 0.f};
  v8f acc[2][4];
#pragma unroll
  for (int mt = 0; mt < 2; ++mt)
#pragma unroll
    for (int nt = 0; nt < 4; ++nt) acc[mt][nt] = z;

  for (int kt = 0; kt < 16; ++kt) {
    int kb = kt * 32;
    int kA = kb + half * 16;
    ABu a0, a1;
    a0.h[0] = *(const bf16x8*)&Xs[l16][kA];
    a0.h[1] = *(const bf16x8*)&Xs[l16][kA + 8];
    a1.h[0] = *(const bf16x8*)&Xs[16 + l16][kA];
    a1.h[1] = *(const bf16x8*)&Xs[16 + l16][kA + 8];
    int kB = kb + half * 16;
    if (kt + 1 < 16)
      __builtin_prefetch(&Wt[(size_t)(w * 64 + l16) * Cn + kB + 32], 0, 1);
#pragma unroll
    for (int nt = 0; nt < 4; ++nt) {
      int n0 = w * 64 + nt * 16;
      bf16x16 bv = *(const bf16x16*)&Wt[(size_t)(n0 + l16) * Cn + kB];
      acc[0][nt] = wmma_bf16(a0.v, bv, acc[0][nt]);
      acc[1][nt] = wmma_bf16(a1.v, bv, acc[1][nt]);
    }
  }

  // per-row online logsumexp over this wave's 64-column strip + column extraction
  float lmx[2][8], lsm[2][8];
#pragma unroll
  for (int mt = 0; mt < 2; ++mt)
#pragma unroll
    for (int i = 0; i < 8; ++i) { lmx[mt][i] = -3.0e38f; lsm[mt][i] = 0.f; }

#pragma unroll
  for (int nt = 0; nt < 4; ++nt) {
    int n = w * 64 + nt * 16 + l16;
    float bias = jbias[n];
#pragma unroll
    for (int mt = 0; mt < 2; ++mt) {
#pragma unroll
      for (int i = 0; i < 8; ++i) {
        float val = acc[mt][nt][i] + bias;
        int m = mt * 16 + half * 8 + i;
        if (n == symc[m]) pxl[m] = val;   // unique writer per m
        if (n == 0)       pyl[m] = val;   // blank column
        float om = lmx[mt][i];
        float nm = fmaxf(om, val);
        lsm[mt][i] = lsm[mt][i] * __expf(om - nm) + __expf(val - nm);
        lmx[mt][i] = nm;
      }
    }
  }
  // reduce (max,sum) across the 16 lanes of each half-wave (rows differ per half)
#pragma unroll
  for (int mt = 0; mt < 2; ++mt) {
#pragma unroll
    for (int i = 0; i < 8; ++i) {
      float mx = lmx[mt][i], sm = lsm[mt][i];
#pragma unroll
      for (int off = 8; off >= 1; off >>= 1) {
        float omx = __shfl_xor(mx, off, 16);
        float osm = __shfl_xor(sm, off, 16);
        float nm = fmaxf(mx, omx);
        sm = sm * __expf(mx - nm) + osm * __expf(omx - nm);
        mx = nm;
      }
      if (l16 == 0) {
        int m = mt * 16 + half * 8 + i;
        wmx[w][m] = mx; wsm[w][m] = sm;
      }
    }
  }
  __syncthreads();
  if (tid < 32) {
    int m = tid;
    float MX = -3.0e38f, SM = 0.f;
    for (int ww = 0; ww < 8; ++ww) {
      float omx = wmx[ww][m], osm = wsm[ww][m];
      float nm = fmaxf(MX, omx);
      SM = SM * __expf(MX - nm) + osm * __expf(omx - nm);
      MX = nm;
    }
    if (m < Rn) {
      float lse = MX + logf(SM);
      int o = bt * Rn + m;
      pxband[o] = pxl[m] - lse;
      pyband[o] = pyl[m] - lse;
    }
  }
}

// ---------------- K10/K11: scatter band into full (NEG-filled) px2/py2 ----------------
__global__ void fillneg_kernel(float* __restrict__ a, int n) {
  int i = blockIdx.x * 256 + threadIdx.x;
  if (i < n) a[i] = NEGF;
}

__global__ void scatter_kernel(const float* __restrict__ pxband, const float* __restrict__ pyband,
                               const int* __restrict__ sbeg,
                               float* __restrict__ px2f, float* __restrict__ py2f) {
  int idx = blockIdx.x * 256 + threadIdx.x;
  if (idx >= Bn * Tn * Rn) return;
  int m = idx % Rn, bt = idx / Rn;
  int t = bt & 511, b = bt >> 9;
  int s = sbeg[bt] + m;                 // <= 128
  if (s <= Sn) py2f[(size_t)(b * Sp1 + s) * Tn + t] = pyband[idx];
  if (s < Sn)  px2f[(size_t)(b * Sn + s) * T1n + t] = pxband[idx];
}

// ---------------- K12: final loss combine ----------------
__global__ void final_kernel(const float* __restrict__ plast, const float* __restrict__ p2last,
                             float* __restrict__ out) {
  if (blockIdx.x == 0 && threadIdx.x == 0) {
    float s1 = 0.f, s2 = 0.f;
    for (int b = 0; b < Bn; ++b) { s1 += plast[b]; s2 += p2last[b]; }
    out[0] = -0.1f * (s1 / (float)Bn) - (s2 / (float)Bn);
  }
}

// ======================= host launcher =======================
extern "C" void kernel_launch(void* const* d_in, const int* in_sizes, int n_in,
                              void* d_out, int out_size, void* d_ws, size_t ws_size,
                              hipStream_t stream) {
  (void)in_sizes; (void)n_in; (void)out_size; (void)ws_size;
  const float* am      = (const float*)d_in[0];  // (B,T,C)
  const float* lm      = (const float*)d_in[1];  // (B,S+1,C)
  const int*   targets = (const int*)d_in[2];    // (B,S)
  /* d_in[3] = y_lens (all == S, unused) */
  const float* jw      = (const float*)d_in[4];  // (C,C)
  const float* jb      = (const float*)d_in[5];  // (C,)
  float* out = (float*)d_out;

  char* w = (char*)d_ws; size_t off = 0;
  auto alloc = [&](size_t bytes) -> void* {
    void* p = w + off;
    off += (bytes + 255) & ~(size_t)255;
    return p;
  };
  __bf16* lm_e   = (__bf16*)alloc((size_t)Bn * Sp1 * Cn * 2);
  __bf16* am_e   = (__bf16*)alloc((size_t)Bn * Tn  * Cn * 2);
  __bf16* Wt     = (__bf16*)alloc((size_t)Cn * Cn * 2);
  float* lm_max  = (float*)alloc((size_t)Bn * Sp1 * 4);
  float* lm_lse  = (float*)alloc((size_t)Bn * Sp1 * 4);
  float* am_max  = (float*)alloc((size_t)Bn * Tn * 4);
  float* nrm     = (float*)alloc((size_t)Bn * Sp1 * Tn * 4);
  float* px1     = (float*)alloc((size_t)Bn * Sn * T1n * 4);
  float* py1     = (float*)alloc((size_t)Bn * Sp1 * Tn * 4);
  float* pmat    = (float*)alloc((size_t)Bn * Sp1 * T1n * 4);
  float* qmat    = (float*)alloc((size_t)Bn * Sp1 * T1n * 4);
  float* gpx     = (float*)alloc((size_t)Bn * Sn * T1n * 4);
  float* gpy     = (float*)alloc((size_t)Bn * Sp1 * Tn * 4);
  int*   sbeg    = (int*)alloc((size_t)Bn * Tn * 4);
  float* pxband  = (float*)alloc((size_t)Bn * Tn * Rn * 4);
  float* pyband  = (float*)alloc((size_t)Bn * Tn * Rn * 4);
  float* px2f    = (float*)alloc((size_t)Bn * Sn * T1n * 4);
  float* py2f    = (float*)alloc((size_t)Bn * Sp1 * Tn * 4);
  float* p2mat   = (float*)alloc((size_t)Bn * Sp1 * T1n * 4);
  float* plast   = (float*)alloc((size_t)Bn * 4);
  float* p2last  = (float*)alloc((size_t)Bn * 4);

  // --- stage 1: row stats + bf16 exp tables, weight transpose ---
  rowstats_kernel<<<Bn * Sp1, 256, 0, stream>>>(lm, lm_e, lm_max, lm_lse);
  rowstats_kernel<<<Bn * Tn,  256, 0, stream>>>(am, am_e, am_max, nullptr);
  wtprep_kernel<<<Cn, 256, 0, stream>>>(jw, Wt);

  // --- stage 2: normalizers via bf16 WMMA, smoothed px/py ---
  normalizer_kernel<<<(Bn * 9 * 32) / 8, 256, 0, stream>>>(lm_e, am_e, lm_max, am_max, nrm);
  smoothed_kernel<<<Bn * Sp1, 256, 0, stream>>>(am, lm, targets, nrm, lm_lse, px1, py1);

  // --- stage 3: simple-loss DP forward/backward (occupancy grads) ---
  mi_forward_kernel<<<Bn, 160, 0, stream>>>(px1, py1, pmat, plast);
  mi_backward_kernel<<<Bn, 160, 0, stream>>>(px1, py1, pmat, plast, qmat, gpx, gpy);

  // --- stage 4: pruning ranges ---
  sbegin_kernel<<<(Bn * Tn + 255) / 256, 256, 0, stream>>>(gpx, gpy, sbeg);
  adjust_kernel<<<1, 32, 0, stream>>>(sbeg);

  // --- stage 5: fused joiner GEMM + logsumexp + gathers (the 41-GFLOP WMMA stage) ---
  joiner_kernel<<<Bn * Tn, 256, 0, stream>>>(am, lm, targets, Wt, jb, sbeg, pxband, pyband);

  // --- stage 6: scatter band, pruned-loss DP, combine ---
  int n1 = Bn * Sn * T1n, n2 = Bn * Sp1 * Tn;
  fillneg_kernel<<<(n1 + 255) / 256, 256, 0, stream>>>(px2f, n1);
  fillneg_kernel<<<(n2 + 255) / 256, 256, 0, stream>>>(py2f, n2);
  scatter_kernel<<<(Bn * Tn * Rn + 255) / 256, 256, 0, stream>>>(pxband, pyband, sbeg, px2f, py2f);
  mi_forward_kernel<<<Bn, 160, 0, stream>>>(px2f, py2f, p2mat, p2last);
  final_kernel<<<1, 32, 0, stream>>>(plast, p2last, out);
}
